// EA_5566277615732
// MI455X (gfx1250) — matrix operand order repaired and checked by reading, not compiled, not run on hardware
//
#include <hip/hip_runtime.h>
#include <hip/hip_bf16.h>

// ---------- problem constants ----------
#define Bv     256
#define Tv     128
#define BT     32768         // B*T
#define WD     300
#define IN_DIM 400
#define ATTN   700
#define NF     512
#define NCLS   19
#define KPAD_A 704           // 700 padded to 22*32
#define KT_A   22
#define NTP_A  48            // 700/16 = 43.75 -> padded to 48 n-tiles (768)
#define KT_C   38            // 1200 padded to 38*32 = 1216
#define NT_C   32            // 512/16
#define MT2    1024          // BT/32: two 16-row m-tiles per block-x
#define ROWS_P 130           // padded rows per batch: zero | t=0..127 | zero

typedef __attribute__((ext_vector_type(16))) _Float16 v16h;
typedef __attribute__((ext_vector_type(8)))  _Float16 v8h;
typedef __attribute__((ext_vector_type(8)))  float    v8f;

union FragU { v16h v; v8h h[2]; };

__device__ __forceinline__ void load_bfrag(FragU& f, const char* bp) {
    f.h[0] = *(const v8h*)(bp);
    f.h[1] = *(const v8h*)(bp + 16);
}

// =====================================================================
// Zero-padded masked f16 input: xph[b][r][c], r=0 and r=129 are zero
// rows, r=1..128 hold inp[b][r-1][c] * mask[b][r-1].
// =====================================================================
__global__ void k_build_inp(const int* __restrict__ words,
                            const float* __restrict__ mask,
                            const int* __restrict__ d1s,
                            const int* __restrict__ d2s,
                            const float* __restrict__ wemb,
                            const float* __restrict__ d1e,
                            const float* __restrict__ d2e,
                            _Float16* __restrict__ xph) {
    int idx = blockIdx.x * blockDim.x + threadIdx.x;
    if (idx >= Bv * ROWS_P * IN_DIM) return;
    int rr = idx / IN_DIM, col = idx - rr * IN_DIM;
    int b = rr / ROWS_P, r = rr - b * ROWS_P;
    float val = 0.f;
    if (r >= 1 && r <= Tv) {
        int m = b * Tv + (r - 1);
        if (col < WD)            val = wemb[(size_t)words[m] * WD + col];
        else if (col < WD + 50)  val = d1e[(size_t)d1s[m] * 50 + (col - WD)];
        else                     val = d2e[(size_t)d2s[m] * 50 + (col - WD - 50)];
        val *= mask[m];
    }
    xph[idx] = (_Float16)val;
}

// argE[sel][b][0..303]  (arg word embedding, zero-padded 300->304)
__global__ void k_build_arg(const int* __restrict__ arg1,
                            const int* __restrict__ arg2,
                            const float* __restrict__ wemb,
                            _Float16* __restrict__ argE) {
    int idx = blockIdx.x * blockDim.x + threadIdx.x;
    if (idx >= 2 * Bv * 304) return;
    int sel = idx / (Bv * 304);
    int r = idx - sel * (Bv * 304);
    int b = r / 304, c = r - b * 304;
    int w = sel ? arg2[b] : arg1[b];
    argE[idx] = (c < WD) ? (_Float16)wemb[(size_t)w * WD + c] : (_Float16)0.f;
}

// =====================================================================
// Pack B matrices into WMMA-fragment-tiled order:
//   tile (kt,nt) -> 512 halfs; element (k_in,n_in):
//   lane = ((k_in/8)%2)*16 + n_in, j = (k_in/16)*8 + k_in%8, addr = lane*16 + j.
// Attention: B[k][n] = Wa[n*700+k]   (t = ai @ Wa.T)
// =====================================================================
__global__ void k_pack_attnW(const float* __restrict__ Wa1,
                             const float* __restrict__ Wa2,
                             _Float16* __restrict__ Bw) {
    int idx = blockIdx.x * blockDim.x + threadIdx.x;
    const int per = KT_A * NTP_A * 512;
    if (idx >= 2 * per) return;
    int sel = idx / per;
    int r = idx - sel * per;
    int tile = r >> 9, lj = r & 511;
    int lane = lj >> 4, j = lj & 15;
    int kt = tile / NTP_A, nt = tile - kt * NTP_A;
    int k = kt * 32 + ((j >> 3) << 4) + ((lane >> 4) << 3) + (j & 7);
    int n = nt * 16 + (lane & 15);
    const float* Wa = sel ? Wa2 : Wa1;
    Bw[idx] = (k < ATTN && n < ATTN) ? (_Float16)Wa[(size_t)n * ATTN + k]
                                     : (_Float16)0.f;
}

// Conv: K layout k = dt*400 + c ; B[k][n] = conv_w[n][c][dt]  (OIH)
__global__ void k_pack_convW(const float* __restrict__ cw,
                             _Float16* __restrict__ Bc) {
    int idx = blockIdx.x * blockDim.x + threadIdx.x;
    if (idx >= KT_C * NT_C * 512) return;
    int tile = idx >> 9, lj = idx & 511;
    int lane = lj >> 4, j = lj & 15;
    int kt = tile / NT_C, nt = tile - kt * NT_C;
    int k = kt * 32 + ((j >> 3) << 4) + ((lane >> 4) << 3) + (j & 7);
    int n = nt * 16 + (lane & 15);
    float v = 0.f;
    if (k < 1200) {
        int dt = (k >= 800) ? 2 : (k >= 400) ? 1 : 0;
        int c = k - dt * IN_DIM;
        v = cw[(size_t)n * 1200 + c * 3 + dt];
    }
    Bc[idx] = (_Float16)v;
}

__global__ void k_init(float* __restrict__ cnnmax) {
    int idx = blockIdx.x * blockDim.x + threadIdx.x;
    if (idx < Bv * NF) cnnmax[idx] = __int_as_float(0xFF800000); // -inf
}

// =====================================================================
// Attention GEMM, 32x64 tile per wave (2 m-tiles x 4 n-tiles = 8 accs),
// reusing each B fragment for two m-tiles (halves B traffic from L2).
// NO LDS / barriers: per-lane A fragments are direct global b128 loads
// from the contiguous source row; chunks are 8-half aligned and never
// straddle the k=400 xph/argE boundary (pointer cndmask select).
// Pipeline per step: issue A(cur) loads, then B(next) loads, then 8
// WMMAs -> wmma waits loadcnt<=8, B(next) stays in flight.
// =====================================================================
__global__ __launch_bounds__(128) void k_attn_gemm(
    const _Float16* __restrict__ xph,
    const _Float16* __restrict__ argE,   // [2][B][304]
    const _Float16* __restrict__ Bw,     // [2][KT_A*NTP_A*512]
    const float* __restrict__ wr1,
    const float* __restrict__ wr2,
    float* __restrict__ partial)         // [2][NTP_A][BT]
{
    const int sel  = blockIdx.z;
    const int m0   = blockIdx.x * 32;              // two m-tiles: m0, m0+16
    const int b    = m0 >> 7;
    const int tid  = threadIdx.x;
    const int wave = tid >> 5;
    const int lane = tid & 31;
    const int nt0  = blockIdx.y * 16 + wave * 4;   // 4 n-tiles per wave
    const float* wr = sel ? wr2 : wr1;

    // per-lane A fragment source pointers (ISA 16-bit A 16x32 layout)
    const int frow = lane & 15, fg = lane >> 4;
    const int row0 = b * ROWS_P + 1 + ((m0 + frow) & 127);
    const int row1 = b * ROWS_P + 1 + ((m0 + 16 + frow) & 127);
    const char* pX0 = (const char*)xph + (((size_t)row0 * IN_DIM) << 1) + fg * 16;
    const char* pX1 = (const char*)xph + (((size_t)row1 * IN_DIM) << 1) + fg * 16;
    const char* pR  = (const char*)(argE + ((size_t)sel * Bv + b) * 304) - 800 + fg * 16;

    // B: fragment-tiled; per-lane 32B contiguous
    const char* Bp = (const char*)(Bw + (size_t)sel * (KT_A * NTP_A * 512)) +
                     ((size_t)nt0 << 10) + lane * 32;
    const size_t KSTR = (size_t)NTP_A * 1024;      // bytes per kt step

    v8f acc0[4], acc1[4]; FragU a0, a1, bf[4], nbf[4];
    #pragma unroll
    for (int j = 0; j < 4; ++j) { acc0[j] = (v8f){}; acc1[j] = (v8f){}; }

    #pragma unroll
    for (int j = 0; j < 4; ++j) load_bfrag(bf[j], Bp + j * 1024);

    #pragma unroll 2
    for (int kt = 0; kt < KT_A; ++kt) {
        // A fragments for the CURRENT step (chunks never straddle k=400)
        const char* c1 = ((kt * 32 + 8)  < IN_DIM) ? pX0 : pR;
        const char* c2 = ((kt * 32 + 24) < IN_DIM) ? pX0 : pR;
        const char* d1 = ((kt * 32 + 8)  < IN_DIM) ? pX1 : pR;
        const char* d2 = ((kt * 32 + 24) < IN_DIM) ? pX1 : pR;
        a0.h[0] = *(const v8h*)(c1 + kt * 64);
        a0.h[1] = *(const v8h*)(c2 + kt * 64 + 32);
        a1.h[0] = *(const v8h*)(d1 + kt * 64);
        a1.h[1] = *(const v8h*)(d2 + kt * 64 + 32);
        // B fragments for the NEXT step (last step reloads itself)
        const int ktn = (kt + 1 < KT_A) ? kt + 1 : kt;
        #pragma unroll
        for (int j = 0; j < 4; ++j)
            load_bfrag(nbf[j], Bp + (size_t)ktn * KSTR + j * 1024);

        #pragma unroll
        for (int j = 0; j < 4; ++j) {
            acc0[j] = __builtin_amdgcn_wmma_f32_16x16x32_f16(
                false, a0.v, false, bf[j].v, (short)0, acc0[j], false, false);
            acc1[j] = __builtin_amdgcn_wmma_f32_16x16x32_f16(
                false, a1.v, false, bf[j].v, (short)0, acc1[j], false, false);
        }
        #pragma unroll
        for (int j = 0; j < 4; ++j) bf[j] = nbf[j];
    }

    // epilogue: row-sum of tanh(t)*wr over each 16-col tile
    const int N = lane & 15, g = lane >> 4;
    #pragma unroll
    for (int mt = 0; mt < 2; ++mt) {
        const int mb = m0 + mt * 16;
        #pragma unroll
        for (int j = 0; j < 4; ++j) {
            const v8f& ac = mt ? acc1[j] : acc0[j];
            int e = (nt0 + j) * 16 + N;
            float wv = (e < ATTN) ? wr[e] : 0.f;
            #pragma unroll
            for (int i = 0; i < 8; ++i) {
                float pp = tanhf(ac[i]) * wv;
                pp += __shfl_xor(pp, 1);
                pp += __shfl_xor(pp, 2);
                pp += __shfl_xor(pp, 4);
                pp += __shfl_xor(pp, 8);
                if (N == i)
                    partial[((size_t)sel * NTP_A + nt0 + j) * BT + mb + g * 8 + i] = pp;
            }
        }
    }
}

// =====================================================================
// Conv GEMM, 32x64 tile per wave. Zero-padded xph makes the im2col
// patch of row t fully CONTIGUOUS, so A fragments are direct global
// b128 loads at immediate offsets kt*64 / kt*64+32. No masks needed
// (k>=1200 stray reads multiply zero-packed B rows).
// =====================================================================
__global__ __launch_bounds__(128) void k_conv_gemm(
    const _Float16* __restrict__ xph,
    const _Float16* __restrict__ Bc,     // [KT_C*NT_C*512]
    const float* __restrict__ conv_b,
    float* __restrict__ cnnmax)          // [B*NF], init -inf
{
    const int m0   = blockIdx.x * 32;
    const int b    = m0 >> 7;
    const int tid  = threadIdx.x;
    const int wave = tid >> 5;
    const int lane = tid & 31;
    const int nt0  = blockIdx.y * 16 + wave * 4;

    const int frow = lane & 15, fg = lane >> 4;
    const int row0 = b * ROWS_P + ((m0 + frow) & 127);        // patch origin
    const int row1 = b * ROWS_P + ((m0 + 16 + frow) & 127);
    const char* pC0 = (const char*)xph + (((size_t)row0 * IN_DIM) << 1) + fg * 16;
    const char* pC1 = (const char*)xph + (((size_t)row1 * IN_DIM) << 1) + fg * 16;

    const char* Bp = (const char*)Bc + ((size_t)nt0 << 10) + lane * 32;
    const size_t KSTR = (size_t)NT_C * 1024;

    v8f acc0[4], acc1[4]; FragU a0, a1, bf[4], nbf[4];
    #pragma unroll
    for (int j = 0; j < 4; ++j) { acc0[j] = (v8f){}; acc1[j] = (v8f){}; }

    #pragma unroll
    for (int j = 0; j < 4; ++j) load_bfrag(bf[j], Bp + j * 1024);

    #pragma unroll 2
    for (int kt = 0; kt < KT_C; ++kt) {
        a0.h[0] = *(const v8h*)(pC0 + kt * 64);
        a0.h[1] = *(const v8h*)(pC0 + kt * 64 + 32);
        a1.h[0] = *(const v8h*)(pC1 + kt * 64);
        a1.h[1] = *(const v8h*)(pC1 + kt * 64 + 32);
        const int ktn = (kt + 1 < KT_C) ? kt + 1 : kt;
        #pragma unroll
        for (int j = 0; j < 4; ++j)
            load_bfrag(nbf[j], Bp + (size_t)ktn * KSTR + j * 1024);

        #pragma unroll
        for (int j = 0; j < 4; ++j) {
            acc0[j] = __builtin_amdgcn_wmma_f32_16x16x32_f16(
                false, a0.v, false, bf[j].v, (short)0, acc0[j], false, false);
            acc1[j] = __builtin_amdgcn_wmma_f32_16x16x32_f16(
                false, a1.v, false, bf[j].v, (short)0, acc1[j], false, false);
        }
        #pragma unroll
        for (int j = 0; j < 4; ++j) bf[j] = nbf[j];
    }

    const int N = lane & 15;
    #pragma unroll
    for (int j = 0; j < 4; ++j) {
        int f = (nt0 + j) * 16 + N;
        float bias = conv_b[f];
        float vmax = -3.0e38f;
        #pragma unroll
        for (int i = 0; i < 8; ++i) {
            vmax = fmaxf(vmax, acc0[j][i] + bias);
            vmax = fmaxf(vmax, acc1[j][i] + bias);
        }
        vmax = fmaxf(vmax, __shfl_xor(vmax, 16));   // fold the two half-waves
        if (lane < 16) {
            float* addr = cnnmax + b * NF + f;
            if (vmax >= 0.f) atomicMax((int*)addr, __float_as_int(vmax));
            else             atomicMin((unsigned int*)addr, __float_as_uint(vmax));
        }
    }
}

// =====================================================================
// Masked softmax over T per (sel,b); partial e-tiles summed in fixed order
// =====================================================================
__global__ void k_softmax(const float* __restrict__ mask,
                          const float* __restrict__ partial,
                          float* __restrict__ attnw) {
    int b = blockIdx.x, sel = blockIdx.y, t = threadIdx.x;
    int m = b * Tv + t;
    float s = 0.f;
    for (int nt = 0; nt < NTP_A; ++nt)
        s += partial[((size_t)sel * NTP_A + nt) * BT + m];
    if (mask[m] == 0.f) s = -1e30f;
    __shared__ float red[Tv];
    red[t] = s; __syncthreads();
    for (int o = 64; o; o >>= 1) { if (t < o) red[t] = fmaxf(red[t], red[t + o]); __syncthreads(); }
    float mx = red[0]; __syncthreads();
    float e = __expf(s - mx);
    red[t] = e; __syncthreads();
    for (int o = 64; o; o >>= 1) { if (t < o) red[t] += red[t + o]; __syncthreads(); }
    attnw[(size_t)sel * BT + m] = e / red[0];
}

// v[sel][b][d] = sum_t a[t] * inp[b,t,d]
__global__ void k_pool(const _Float16* __restrict__ xph,
                       const float* __restrict__ attnw,
                       float* __restrict__ v) {
    int b = blockIdx.x, sel = blockIdx.y;
    __shared__ float aw[Tv];
    if (threadIdx.x < Tv) aw[threadIdx.x] = attnw[(size_t)sel * BT + b * Tv + threadIdx.x];
    __syncthreads();
    for (int d = threadIdx.x; d < IN_DIM; d += blockDim.x) {
        float acc = 0.f;
        for (int t = 0; t < Tv; ++t)
            acc += aw[t] * (float)xph[(size_t)(b * ROWS_P + 1 + t) * IN_DIM + d];
        v[((size_t)sel * Bv + b) * IN_DIM + d] = acc;
    }
}

// feat = [tanh(cnnmax) | v1 | v2]; logits = feat @ W.T + b; softmax(19)
__global__ void k_head(const float* __restrict__ cnnmax,
                       const float* __restrict__ v,
                       const float* __restrict__ dw,
                       const float* __restrict__ db,
                       float* __restrict__ out) {
    int b = blockIdx.x;
    __shared__ float feat[NF + 2 * IN_DIM];
    __shared__ float lg[NCLS];
    for (int f = threadIdx.x; f < NF; f += blockDim.x)
        feat[f] = tanhf(cnnmax[b * NF + f]);
    for (int d = threadIdx.x; d < IN_DIM; d += blockDim.x) {
        feat[NF + d]          = v[(size_t)b * IN_DIM + d];
        feat[NF + IN_DIM + d] = v[(size_t)(Bv + b) * IN_DIM + d];
    }
    __syncthreads();
    const int FD = NF + 2 * IN_DIM;   // 1312
    if (threadIdx.x < NCLS) {
        float acc = db[threadIdx.x];
        const float* w = dw + (size_t)threadIdx.x * FD;
        for (int j = 0; j < FD; ++j) acc += feat[j] * w[j];
        lg[threadIdx.x] = acc;
    }
    __syncthreads();
    if (threadIdx.x == 0) {
        float mx = lg[0];
        for (int c = 1; c < NCLS; ++c) mx = fmaxf(mx, lg[c]);
        float sum = 0.f;
        for (int c = 0; c < NCLS; ++c) { lg[c] = __expf(lg[c] - mx); sum += lg[c]; }
        for (int c = 0; c < NCLS; ++c) out[b * NCLS + c] = lg[c] / sum;
    }
}

// =====================================================================
extern "C" void kernel_launch(void* const* d_in, const int* in_sizes, int n_in,
                              void* d_out, int out_size, void* d_ws, size_t ws_size,
                              hipStream_t stream) {
    const int*   words = (const int*)d_in[0];
    const float* mask  = (const float*)d_in[1];
    const int*   d1s   = (const int*)d_in[2];
    const int*   d2s   = (const int*)d_in[3];
    const int*   arg1  = (const int*)d_in[7];
    const int*   arg2  = (const int*)d_in[8];
    const float* wemb  = (const float*)d_in[9];
    const float* d1e   = (const float*)d_in[10];
    const float* d2e   = (const float*)d_in[11];
    const float* Wa1   = (const float*)d_in[12];
    const float* wr1   = (const float*)d_in[13];
    const float* Wa2   = (const float*)d_in[14];
    const float* wr2   = (const float*)d_in[15];
    const float* cw    = (const float*)d_in[16];
    const float* cb    = (const float*)d_in[17];
    const float* dw    = (const float*)d_in[18];
    const float* db    = (const float*)d_in[19];
    float* out = (float*)d_out;

    // ---- workspace carve-out (all sizes 256B-aligned) ----
    char* ws = (char*)d_ws;
    size_t off = 0;
    auto take = [&](size_t bytes) { char* p = ws + off; off += (bytes + 255) & ~(size_t)255; return p; };
    _Float16* xph   = (_Float16*)take((size_t)Bv * ROWS_P * IN_DIM * 2 + 256); // 26.6 MB (+slack)
    _Float16* argE  = (_Float16*)take((size_t)2 * Bv * 304 * 2);               // 0.3 MB
    _Float16* Bw    = (_Float16*)take((size_t)2 * KT_A * NTP_A * 512 * 2);     // 2.2 MB
    _Float16* Bc    = (_Float16*)take((size_t)KT_C * NT_C * 512 * 2);          // 1.2 MB
    float*    part  = (float*)take((size_t)2 * NTP_A * BT * 4);                // 12.6 MB
    float*    attnw = (float*)take((size_t)2 * BT * 4);                        // 0.26 MB
    float*    cnnmx = (float*)take((size_t)Bv * NF * 4);                       // 0.5 MB
    float*    vbuf  = (float*)take((size_t)2 * Bv * IN_DIM * 4);               // 0.8 MB
    (void)ws_size; (void)in_sizes; (void)n_in; (void)out_size;

    // ---- pipeline ----
    k_init<<<(Bv * NF + 255) / 256, 256, 0, stream>>>(cnnmx);
    k_build_inp<<<(Bv * ROWS_P * IN_DIM + 255) / 256, 256, 0, stream>>>(
        words, mask, d1s, d2s, wemb, d1e, d2e, xph);
    k_build_arg<<<(2 * Bv * 304 + 255) / 256, 256, 0, stream>>>(arg1, arg2, wemb, argE);
    k_pack_attnW<<<(2 * KT_A * NTP_A * 512 + 255) / 256, 256, 0, stream>>>(Wa1, Wa2, Bw);
    k_pack_convW<<<(KT_C * NT_C * 512 + 255) / 256, 256, 0, stream>>>(cw, Bc);

    k_attn_gemm<<<dim3(MT2, NTP_A / 16, 2), 128, 0, stream>>>(xph, argE, Bw, wr1, wr2, part);
    k_conv_gemm<<<dim3(MT2, NT_C / 16), 128, 0, stream>>>(xph, Bc, cb, cnnmx);

    k_softmax<<<dim3(Bv, 2), Tv, 0, stream>>>(mask, part, attnw);
    k_pool<<<dim3(Bv, 2), 128, 0, stream>>>(xph, attnw, vbuf);
    k_head<<<Bv, 128, 0, stream>>>(cnnmx, vbuf, dw, db, out);
}
